// VectorQuantizer_12275016532126
// MI455X (gfx1250) — matrix operand order repaired and checked by reading, not compiled, not run on hardware
//
#include <hip/hip_runtime.h>
#include <hip/hip_bf16.h>

typedef __attribute__((ext_vector_type(16))) _Float16 v16h;
typedef __attribute__((ext_vector_type(8)))  float    v8f;

constexpr int Bb = 32, Dd = 64, Hh = 32, Ww = 32;
constexpr int Nn = Bb * Hh * Ww;   // 32768 positions
constexpr int Kk = 1024;           // codebook size

// ---------------- init: zero accumulators (ws is poisoned 0xAA) ----------------
__global__ void vq_init(int* __restrict__ counts, float* __restrict__ lossacc) {
    int t = threadIdx.x;
    for (int i = t; i < Kk; i += 256) counts[i] = 0;
    if (t == 0) *lossacc = 0.f;
}

// ------------- normalize inputs, transpose [B,D,H,W] -> row-major [N,64] f16 ---
__global__ void vq_norm_inputs(const float* __restrict__ in, _Float16* __restrict__ A) {
    int n = blockIdx.x * 256 + threadIdx.x;     // n = b*1024 + h*32 + w
    int b = n >> 10, hw = n & 1023;
    const float* p = in + (size_t)b * (Dd * Hh * Ww) + hw;   // coalesced over hw per d
    float v[Dd];
    float ss = 0.f;
#pragma unroll
    for (int d = 0; d < Dd; ++d) { float x = p[d * 1024]; v[d] = x; ss += x * x; }
    float nr = fmaxf(sqrtf(ss), 1e-12f);
    float s = 1.f / nr;
    union { _Float16 h[Dd]; uint4 u[8]; } u;
#pragma unroll
    for (int d = 0; d < Dd; ++d) u.h[d] = (_Float16)(v[d] * s);
    uint4* o = (uint4*)(A + (size_t)n * Dd);
#pragma unroll
    for (int i = 0; i < 8; ++i) o[i] = u.u[i];
}

// ------------- normalize codebook rows -> row-major [K,64] f16 -----------------
__global__ void vq_norm_weights(const float* __restrict__ w, _Float16* __restrict__ Wn) {
    int k = blockIdx.x * 256 + threadIdx.x;
    const float* p = w + (size_t)k * Dd;
    float v[Dd];
    float ss = 0.f;
#pragma unroll
    for (int d = 0; d < Dd; ++d) { float x = p[d]; v[d] = x; ss += x * x; }
    float nr = fmaxf(sqrtf(ss), 1e-12f);
    float s = 1.f / nr;
    union { _Float16 h[Dd]; uint4 u[8]; } u;
#pragma unroll
    for (int d = 0; d < Dd; ++d) u.h[d] = (_Float16)(v[d] * s);
    uint4* o = (uint4*)(Wn + (size_t)k * Dd);
#pragma unroll
    for (int i = 0; i < 8; ++i) o[i] = u.u[i];
}

// -------- fused GEMM (WMMA f16) + argmax(cosine) == argmin(distance) -----------
// Block = 256 threads = 8 waves; each wave owns one 16-row tile of N.
// Entire f16 codebook (128 KB) staged into LDS with GLOBAL_LOAD_ASYNC_TO_LDS_B128.
__global__ void __launch_bounds__(256) vq_argmin(const _Float16* __restrict__ A,
                                                 const _Float16* __restrict__ Wn,
                                                 int* __restrict__ idx_out,
                                                 int* __restrict__ counts) {
    __shared__ __align__(16) _Float16 sW[Kk * Dd];   // 131072 bytes

    int tid = threadIdx.x;

    // ---- async codebook staging: 8192 b128 transfers, 32 per thread ----
    // GVS mode: mem = SGPR base + VGPR 32-bit offset; VDST VGPR = LDS byte address.
    {
        unsigned long long gbase = (unsigned long long)Wn;
        unsigned lds0 = (unsigned)(uintptr_t)(&sW[0]) + tid * 16u;
        unsigned gof0 = tid * 16u;
#pragma unroll
        for (int i = 0; i < 32; ++i) {
            unsigned ldsa = lds0 + i * 4096u;
            unsigned gofs = gof0 + i * 4096u;
            asm volatile("global_load_async_to_lds_b128 %0, %1, %2"
                         :: "v"(ldsa), "v"(gofs), "s"(gbase)
                         : "memory");
        }
        asm volatile("s_wait_asynccnt 0x0" ::: "memory");
    }

    int wave = tid >> 5, lane = tid & 31;
    int half = lane >> 4, lr = lane & 15;
    int n0 = (blockIdx.x * 8 + wave) * 16;
    int row = n0 + lr;

    // A fragment layout (ISA 7.12.2, 16-bit A 16x32):
    // lanes 0-15: K={0-7,16-23}; lanes 16-31: same row, K={8-15,24-31}
    const uint4* A4 = (const uint4*)A;
    union F { uint4 u[2]; v16h h; };
    F a_lo, a_hi;
    size_t ab = (size_t)row * 8 + half;
    a_lo.u[0] = A4[ab];     a_lo.u[1] = A4[ab + 2];   // d 0..31
    a_hi.u[0] = A4[ab + 4]; a_hi.u[1] = A4[ab + 6];   // d 32..63

    __syncthreads();

    float bestv[8]; int besti[8];
#pragma unroll
    for (int j = 0; j < 8; ++j) { bestv[j] = -1e30f; besti[j] = 0; }

    const uint4* L4 = (const uint4*)sW;
    // Unroll by 2: WMMAs of tile kc+1 co-execute with the argmax VALU of tile kc,
    // hiding the 4-cycle WMMA->VALU hazard.
    for (int kc = 0; kc < Kk / 16; kc += 2) {
        int r0 = kc * 16 + lr;
        int r1 = r0 + 16;
        F b0_lo, b0_hi, b1_lo, b1_hi;
        int bb0 = r0 * 8 + half;
        int bb1 = r1 * 8 + half;
        b0_lo.u[0] = L4[bb0];     b0_lo.u[1] = L4[bb0 + 2];
        b0_hi.u[0] = L4[bb0 + 4]; b0_hi.u[1] = L4[bb0 + 6];
        b1_lo.u[0] = L4[bb1];     b1_lo.u[1] = L4[bb1 + 2];
        b1_hi.u[0] = L4[bb1 + 4]; b1_hi.u[1] = L4[bb1 + 6];

        v8f c0 = {0.f, 0.f, 0.f, 0.f, 0.f, 0.f, 0.f, 0.f};
        v8f c1 = {0.f, 0.f, 0.f, 0.f, 0.f, 0.f, 0.f, 0.f};
        c0 = __builtin_amdgcn_wmma_f32_16x16x32_f16(false, a_lo.h, false, b0_lo.h,
                                                    (short)0, c0, false, false);
        c0 = __builtin_amdgcn_wmma_f32_16x16x32_f16(false, a_hi.h, false, b0_hi.h,
                                                    (short)0, c0, false, false);
        c1 = __builtin_amdgcn_wmma_f32_16x16x32_f16(false, a_lo.h, false, b1_lo.h,
                                                    (short)0, c1, false, false);
        c1 = __builtin_amdgcn_wmma_f32_16x16x32_f16(false, a_hi.h, false, b1_hi.h,
                                                    (short)0, c1, false, false);

        // C/D layout: VGPR j, lanes 0-15 -> M=j; lanes 16-31 -> M=j+8; N = lane&15.
        int col0 = kc * 16 + lr;
        int col1 = col0 + 16;
#pragma unroll
        for (int j = 0; j < 8; ++j) {
            if (c0[j] > bestv[j]) { bestv[j] = c0[j]; besti[j] = col0; }
        }
#pragma unroll
        for (int j = 0; j < 8; ++j) {
            if (c1[j] > bestv[j]) { bestv[j] = c1[j]; besti[j] = col1; }
        }
    }

    // reduce (max value, min index on tie) across the 16 lanes of each half
#pragma unroll
    for (int j = 0; j < 8; ++j) {
#pragma unroll
        for (int m = 1; m < 16; m <<= 1) {
            float ov = __shfl_xor(bestv[j], m, 16);
            int   oi = __shfl_xor(besti[j], m, 16);
            if (ov > bestv[j] || (ov == bestv[j] && oi < besti[j])) {
                bestv[j] = ov; besti[j] = oi;
            }
        }
    }
    if (lr == 0) {
#pragma unroll
        for (int j = 0; j < 8; ++j) {
            int rw = n0 + half * 8 + j;
            idx_out[rw] = besti[j];
            atomicAdd(&counts[besti[j]], 1);
        }
    }
}

// -------- gather codes back to [B,D,H,W] (== quantized_st) + loss partials -----
__global__ void vq_gather_loss(const float* __restrict__ in, const float* __restrict__ emb,
                               const int* __restrict__ idx, float* __restrict__ out,
                               float* __restrict__ lossacc) {
    int n = blockIdx.x * 256 + threadIdx.x;
    int b = n >> 10, hw = n & 1023;
    int k = idx[n];
    const float* e = emb + (size_t)k * Dd;          // L2-resident (256 KB table)
    const float* p = in + (size_t)b * 65536 + hw;
    float* o = out + (size_t)b * 65536 + hw;
    float acc = 0.f;
#pragma unroll
    for (int d = 0; d < Dd; ++d) {
        float q = e[d];
        float x = p[d * 1024];
        o[d * 1024] = q;                            // quantized_st value == quantized
        float df = q - x;
        acc += df * df;
    }
    __shared__ float red[256];
    red[threadIdx.x] = acc;
    __syncthreads();
    for (int s = 128; s > 0; s >>= 1) {
        if (threadIdx.x < s) red[threadIdx.x] += red[threadIdx.x + s];
        __syncthreads();
    }
    if (threadIdx.x == 0) atomicAdd(lossacc, red[0]);
}

// ---------------- scalars: loss = 1.25*mse, perplexity from histogram ----------
__global__ void vq_finalize(const int* __restrict__ counts, const float* __restrict__ lossacc,
                            float* __restrict__ out) {
    __shared__ float red[256];
    int t = threadIdx.x;
    float s = 0.f;
    for (int k = t; k < Kk; k += 256) {
        float p = (float)counts[k] / (float)Nn;
        s += p * logf(p + 1e-10f);
    }
    red[t] = s;
    __syncthreads();
    for (int st = 128; st > 0; st >>= 1) { if (t < st) red[t] += red[t + st]; __syncthreads(); }
    if (t == 0) {
        out[(size_t)Nn * Dd]     = 1.25f * (*lossacc) / (float)((size_t)Nn * Dd);  // q + 0.25*e
        out[(size_t)Nn * Dd + 1] = expf(-red[0]);                                  // perplexity
    }
}

extern "C" void kernel_launch(void* const* d_in, const int* in_sizes, int n_in,
                              void* d_out, int out_size, void* d_ws, size_t ws_size,
                              hipStream_t stream) {
    (void)in_sizes; (void)n_in; (void)out_size; (void)ws_size;
    const float* in  = (const float*)d_in[0];   // [32,64,32,32] f32
    const float* emb = (const float*)d_in[1];   // [1024,64] f32
    float* out = (float*)d_out;                 // [32*64*32*32] + loss + perplexity

    char* ws = (char*)d_ws;
    _Float16* A  = (_Float16*)(ws);                               // 4 MB  [N,64] f16
    _Float16* Wn = (_Float16*)(ws + 4194304);                     // 128 KB [K,64] f16
    int* idx     = (int*)(ws + 4194304 + 131072);                 // 128 KB
    int* counts  = (int*)(ws + 4194304 + 131072 + 131072);        // 4 KB
    float* lossacc = (float*)(ws + 4194304 + 131072 + 131072 + 4096);

    vq_init        <<<1,           256, 0, stream>>>(counts, lossacc);
    vq_norm_inputs <<<Nn / 256,    256, 0, stream>>>(in, A);
    vq_norm_weights<<<Kk / 256,    256, 0, stream>>>(emb, Wn);
    vq_argmin      <<<Nn / (16*8), 256, 0, stream>>>(A, Wn, idx, counts);
    vq_gather_loss <<<Nn / 256,    256, 0, stream>>>(in, emb, idx, out, lossacc);
    vq_finalize    <<<1,           256, 0, stream>>>(counts, lossacc, out);
}
